// EquivariantLocalScoreMachine_62122406969494
// MI455X (gfx1250) — matrix operand order repaired and checked by reading, not compiled, not run on hardware
//
#include <hip/hip_runtime.h>
#include <hip/hip_bf16.h>

typedef __attribute__((ext_vector_type(16))) _Float16 v16h;
typedef __attribute__((ext_vector_type(8)))  float    v8f;

#define Bn   4
#define Cn   3
#define Hn   32
#define Wn   32
#define NQ   4096    // B*H*W queries
#define NP   32768   // N_IMAGES*H*W patches
#define KF   27      // C*9 real features
#define KP   32      // padded K for one 16x16x32 WMMA
#define L2E  1.4426950408889634f

// 16-bit A-fragment swizzle: element position within the 32-half row so that
// lane half 0 reads [K0..7, K16..23] and lane half 1 reads [K8..15, K24..31]
// as one contiguous 32B chunk (ISA 7.12.2, 16-bit A 16x32 layout).
__device__ __forceinline__ int a_pos(int k) {
  return ((k & 8) << 1) + ((k & 16) >> 1) + (k & 7);
}

// ---------------------------------------------------------------------------
// Query-side patch matrix (circular padding), f16, K-contiguous (B-fragment
// order: lane = column, K = half*16 + e). No norms needed: the per-query
// bias cancels in the softmax ratio wc/sum_w.
// ---------------------------------------------------------------------------
__global__ __launch_bounds__(256) void build_q(const float* __restrict__ x,
                                               _Float16* __restrict__ Qm) {
  int q = blockIdx.x * blockDim.x + threadIdx.x;
  if (q >= NQ) return;
  int b = q >> 10, rem = q & 1023, h = rem >> 5, w = rem & 31;
  _Float16* row = Qm + q * KP;
  #pragma unroll
  for (int c = 0; c < Cn; ++c) {
    const float* xc = x + ((b * Cn + c) * Hn) * Wn;
    #pragma unroll
    for (int di = 0; di < 3; ++di) {
      int hh = (h + di - 1 + Hn) & (Hn - 1);
      #pragma unroll
      for (int dj = 0; dj < 3; ++dj) {
        int ww = (w + dj - 1 + Wn) & (Wn - 1);
        row[c * 9 + di * 3 + dj] = (_Float16)xc[hh * Wn + ww];
      }
    }
  }
  #pragma unroll
  for (int k = KF; k < KP; ++k) row[k] = (_Float16)0.f;
}

// ---------------------------------------------------------------------------
// Dataset-side patch matrix (zero padding), f16, stored K-contiguous
// (K = half*16 + e, same order as Qm). Emits the pre-scaled base-2 log-weight bias
//   pb[p] = -(mu^2 / (2 sigma^2)) * log2e * ||patch||^2
// so the hot loop needs a single FMA per score, plus center pixels.
// ---------------------------------------------------------------------------
__global__ __launch_bounds__(256) void build_p(const float* __restrict__ img,
                                               const float* __restrict__ mus,
                                               const float* __restrict__ sgs,
                                               const int* __restrict__ tp,
                                               _Float16* __restrict__ Pm,
                                               float* __restrict__ pb,
                                               float* __restrict__ pc) {
  int p = blockIdx.x * blockDim.x + threadIdx.x;
  if (p >= NP) return;
  const int   t   = *tp;
  const float mu  = mus[t];
  const float sg  = sgs[t];
  const float cpl = mu * mu / (2.f * sg * sg) * L2E;

  int n = p >> 10, rem = p & 1023, h = rem >> 5, w = rem & 31;
  float vals[KP];
  #pragma unroll
  for (int k = KF; k < KP; ++k) vals[k] = 0.f;
  float norm = 0.f;
  #pragma unroll
  for (int c = 0; c < Cn; ++c) {
    const float* ic = img + ((n * Cn + c) * Hn) * Wn;
    #pragma unroll
    for (int di = 0; di < 3; ++di) {
      int hh = h + di - 1;
      #pragma unroll
      for (int dj = 0; dj < 3; ++dj) {
        int ww = w + dj - 1;
        float v = (hh >= 0 && hh < Hn && ww >= 0 && ww < Wn) ? ic[hh * Wn + ww] : 0.f;
        norm += v * v;
        vals[c * 9 + di * 3 + dj] = v;
      }
    }
    pc[p * 3 + c] = ic[h * Wn + w];  // center pixel (k offset 4)
  }
  _Float16* row = Pm + p * KP;
  #pragma unroll
  for (int k = 0; k < KP; ++k) row[k] = (_Float16)vals[k];
  pb[p] = -cpl * norm;
}

// ---------------------------------------------------------------------------
// Fused flash-attention-style score kernel.
//  - Patches stream through the A fragment (M axis), queries sit in a fixed
//    B fragment (N axis): each lane owns exactly ONE query (N = lane&15) and
//    sees 8 patch-scores per WMMA (M = r + 8*(lane>=16)).
//  - Two tiles per iteration: 2 independent WMMAs fill each other's RAW
//    hazard slots; one online-softmax rescale covers all 16 scores.
//  - Base-2 log domain; per-score cost: 1 FMA + 1 max + 1 sub + 1 exp + 4 FMA.
// ---------------------------------------------------------------------------
__global__ __launch_bounds__(256) void score_kernel(
    const _Float16* __restrict__ Qm, const _Float16* __restrict__ Pm,
    const float* __restrict__ pb,    const float* __restrict__ pc,
    const float* __restrict__ x,     const float* __restrict__ mus,
    const float* __restrict__ sgs,   const int* __restrict__ tp,
    float* __restrict__ out) {
  __shared__ float sm[8 * 16 * 5];

  const int   t      = *tp;
  const float mu     = mus[t];
  const float sg     = sgs[t];
  const float inv2s2 = 1.f / (2.f * sg * sg);
  const float csl    = 2.f * mu * inv2s2 * L2E;  // base-2 coeff on dot product

  const int lane  = threadIdx.x & 31;
  const int wave  = threadIdx.x >> 5;
  const int half  = lane >> 4;
  const int l16   = lane & 15;
  const int qbase = blockIdx.x << 4;

  // Fixed B fragment (queries): column N = l16, contiguous K = half*16 + e.
  v16h bfrag = *(const v16h*)(Qm + (qbase + l16) * KP + half * 16);

  // This lane's online softmax state for query (qbase + l16).
  float m = -1e30f, sw = 0.f, a0 = 0.f, a1 = 0.f, a2 = 0.f;

  for (int pt = wave * 2; pt < NP / 16; pt += 16) {  // tiles pt and pt+1
    float lw[2][8];
    float cen[2][24];
    #pragma unroll
    for (int u = 0; u < 2; ++u) {
      const int tile = pt + u;
      // Streaming A fragment: K-contiguous patch rows (K = half*16 + e), contiguous 32B/lane.
      v16h af = *(const v16h*)(Pm + (tile * 16 + l16) * KP + half * 16);
      v8f dz = {};
      v8f d = __builtin_amdgcn_wmma_f32_16x16x32_f16(
          false, af, false, bfrag, (short)0, dz, false, false);

      // This lane's 8 patches for tile u: rows pbase..pbase+7.
      const int pbase = tile * 16 + 8 * half;
      float4 q0 = *(const float4*)(pb + pbase);
      float4 q1 = *(const float4*)(pb + pbase + 4);
      float pbv[8] = {q0.x, q0.y, q0.z, q0.w, q1.x, q1.y, q1.z, q1.w};
      const float4* cp4 = (const float4*)(pc + pbase * 3);  // 24 floats, 16B aligned
      #pragma unroll
      for (int j = 0; j < 6; ++j) {
        float4 cv = cp4[j];
        cen[u][4 * j + 0] = cv.x; cen[u][4 * j + 1] = cv.y;
        cen[u][4 * j + 2] = cv.z; cen[u][4 * j + 3] = cv.w;
      }
      #pragma unroll
      for (int r = 0; r < 8; ++r)
        lw[u][r] = __builtin_fmaf(csl, d[r], pbv[r]);  // single FMA per score
    }

    // One max / one rescale for all 16 scores (v_max3 tree).
    float tm = lw[0][0];
    #pragma unroll
    for (int u = 0; u < 2; ++u)
      #pragma unroll
      for (int r = 0; r < 8; ++r) tm = fmaxf(tm, lw[u][r]);
    float M  = fmaxf(m, tm);
    float e0 = __builtin_amdgcn_exp2f(m - M);
    sw *= e0; a0 *= e0; a1 *= e0; a2 *= e0;
    m = M;
    #pragma unroll
    for (int u = 0; u < 2; ++u)
      #pragma unroll
      for (int r = 0; r < 8; ++r) {
        float e1 = __builtin_amdgcn_exp2f(lw[u][r] - M);
        sw += e1;
        a0 = __builtin_fmaf(e1, cen[u][3 * r + 0], a0);
        a1 = __builtin_fmaf(e1, cen[u][3 * r + 1], a1);
        a2 = __builtin_fmaf(e1, cen[u][3 * r + 2], a2);
      }
  }

  // Merge the two halves (same query, disjoint patch rows): one xor-16 step.
  {
    float m2 = __shfl_xor(m, 16);
    float s2 = __shfl_xor(sw, 16);
    float b0 = __shfl_xor(a0, 16);
    float b1 = __shfl_xor(a1, 16);
    float b2 = __shfl_xor(a2, 16);
    float M  = fmaxf(m, m2);
    float e0 = __builtin_amdgcn_exp2f(m - M);
    float e1 = __builtin_amdgcn_exp2f(m2 - M);
    sw = sw * e0 + s2 * e1;
    a0 = a0 * e0 + b0 * e1;
    a1 = a1 * e0 + b1 * e1;
    a2 = a2 * e0 + b2 * e1;
    m = M;
  }

  if (half == 0) {
    float* pp = &sm[(wave * 16 + l16) * 5];
    pp[0] = m; pp[1] = sw; pp[2] = a0; pp[3] = a1; pp[4] = a2;
  }
  __syncthreads();

  // Merge the 8 wave-partials per query and emit the score.
  if (threadIdx.x < 16) {
    float M = -1e30f, SW = 0.f, W0 = 0.f, W1 = 0.f, W2 = 0.f;
    #pragma unroll
    for (int wv = 0; wv < 8; ++wv) {
      const float* pp = &sm[(wv * 16 + threadIdx.x) * 5];
      float m2 = pp[0];
      float Mn = fmaxf(M, m2);
      float e0 = __builtin_amdgcn_exp2f(M - Mn);
      float e1 = __builtin_amdgcn_exp2f(m2 - Mn);
      SW = SW * e0 + pp[1] * e1;
      W0 = W0 * e0 + pp[2] * e1;
      W1 = W1 * e0 + pp[3] * e1;
      W2 = W2 * e0 + pp[4] * e1;
      M = Mn;
    }
    const int q = qbase + threadIdx.x;
    const int b = q >> 10, rem = q & 1023, h = rem >> 5, w = rem & 31;
    const float invs2 = 1.f / (sg * sg);
    const float invSW = 1.f / SW;
    float wc[3] = {W0, W1, W2};
    #pragma unroll
    for (int c = 0; c < Cn; ++c) {
      int idx = ((b * Cn + c) * Hn + h) * Wn + w;
      // -(x*sum_w - mu*wc) / (sigma^2*sum_w) == -(x - mu*wc/sum_w)/sigma^2
      out[idx] = -(x[idx] - mu * wc[c] * invSW) * invs2;
    }
  }
}

// ---------------------------------------------------------------------------
extern "C" void kernel_launch(void* const* d_in, const int* in_sizes, int n_in,
                              void* d_out, int out_size, void* d_ws, size_t ws_size,
                              hipStream_t stream) {
  (void)in_sizes; (void)n_in; (void)out_size; (void)ws_size;
  const float* x    = (const float*)d_in[0];   // [4,3,32,32]
  const float* imgs = (const float*)d_in[1];   // [32,3,32,32]
  const float* mus  = (const float*)d_in[2];   // [20]
  const float* sgs  = (const float*)d_in[3];   // [20]
  const int*   tp   = (const int*)d_in[4];     // scalar t

  // Workspace layout (bytes):
  //   Qm  f16 [4096 ][32] :       0 .. 262144
  //   Pm  f16 [32768][32] :  262144 .. 2359296   (K-contiguous)
  //   pb  f32 [32768]     : 2359296 .. 2490368   (pre-scaled log2 bias)
  //   pc  f32 [32768][3]  : 2490368 .. 2883584
  char* ws = (char*)d_ws;
  _Float16* Qm = (_Float16*)(ws);
  _Float16* Pm = (_Float16*)(ws + 262144);
  float*    pbp = (float*)  (ws + 2359296);
  float*    pcp = (float*)  (ws + 2490368);

  build_q<<<NQ / 256, 256, 0, stream>>>(x, Qm);
  build_p<<<NP / 256, 256, 0, stream>>>(imgs, mus, sgs, tp, Pm, pbp, pcp);
  score_kernel<<<NQ / 16, 256, 0, stream>>>(Qm, Pm, pbp, pcp, x,
                                            mus, sgs, tp, (float*)d_out);
}
